// BitNetAttention_19421842112749
// MI455X (gfx1250) — matrix operand order repaired and checked by reading, not compiled, not run on hardware
//
#include <hip/hip_runtime.h>
#include <stdint.h>

// ---------------------------------------------------------------------------
// BitNet attention block for MI455X (gfx1250, wave32, WMMA + TDM).
// Sizes fixed by the reference: B=2, S=2048, HID=2560, NH=20, NKV=5, HD=128.
// ---------------------------------------------------------------------------

#define B_      2
#define S_      2048
#define HID_    2560
#define NH_     20
#define NKV_    5
#define HD_     128
#define GROUPS_ 4
#define TOKENS_ (B_ * S_)   // 4096

typedef __bf16 bf16_t;
typedef __attribute__((ext_vector_type(16))) __bf16 v16bf;
typedef __attribute__((ext_vector_type(8)))  __bf16 v8bf;
typedef __attribute__((ext_vector_type(4)))  __bf16 v4bf;
typedef __attribute__((ext_vector_type(8)))  float  v8f;
typedef __attribute__((ext_vector_type(4)))  float  v4f;
typedef __attribute__((ext_vector_type(4)))  unsigned int v4u;
typedef __attribute__((ext_vector_type(4)))  int    v4i;
typedef __attribute__((ext_vector_type(8)))  int    v8i;

#ifndef __has_builtin
#define __has_builtin(x) 0
#endif
#define HAVE_TDM __has_builtin(__builtin_amdgcn_tensor_load_to_lds)

// Concatenate two 8-elem halves into a WMMA operand (register coalescing only).
__device__ __forceinline__ v16bf cat8(v8bf lo, v8bf hi) {
  return __builtin_shufflevector(lo, hi, 0, 1, 2, 3, 4, 5, 6, 7,
                                 8, 9, 10, 11, 12, 13, 14, 15);
}

// bf16 bit pattern for ternary trit t in {0,1,2} -> {-1, 0, +1}; integer-only.
__device__ __forceinline__ uint32_t bfpat(uint32_t t) {
  return (t == 1u) ? 0u : ((t == 0u) ? 0xBF80u : 0x3F80u);
}

// ---------------------------------------------------------------------------
// Ternary-weight GEMM:  Y[M,N] = X[M,K] * unpack(Wp)[N,K]^T * scale
// Packed layout (pack_ternary): per 128-elem block (32 bytes), byte j holds
// trits for k = j, j+32, j+64, j+96 in bit fields [7:6],[5:4],[3:2],[1:0].
// Block: 256 threads = 8 waves (4 M-waves x 2 N-waves); wave tile 32x32.
//
// The raw packed weight tile (64 rows x 32 bytes, row stride Kp) is moved
// global->LDS by the Tensor Data Mover (2D D#: data_size=1B, tile 32x64),
// overlapping the A-tile staging; completion via s_wait_tensorcnt.
//
// Fragment layouts (ISA 7.12.2, 16-bit A 16x32):
//   A: lane = row; VGPR0..3 <- k = 8*half + 0..7 ; VGPR4..7 <- k = 16+8*half+0..7
//   B: lane = n col; element s <- same k as A's element s -> Bs stored TRANSPOSED [k][n].
// ---------------------------------------------------------------------------
#define BM 128
#define BN 64
#define BK 64

__global__ __launch_bounds__(256)
void bitlinear_wmma(const float* __restrict__ X, const uint8_t* __restrict__ Wp,
                    const float* __restrict__ scale_p, float* __restrict__ Y,
                    int M, int K, int N)
{
  __shared__ bf16_t  As[BM][BK + 8];     // row stride 144 B (16B multiple)
  __shared__ bf16_t  Bs[BK][BN + 8];     // TRANSPOSED [k][n]
  __shared__ uint8_t Wstage[BN * 32];    // raw packed weight tile (TDM dest)

  (void)M;
  const int tid  = threadIdx.x;
  const int lane = tid & 31;
  const int wave = tid >> 5;
  const int wm   = wave >> 1;   // 0..3
  const int wn   = wave & 1;    // 0..1
  const int half = lane >> 4;
  const int ml   = lane & 15;
  const int bm0  = blockIdx.y * BM;
  const int bn0  = blockIdx.x * BN;
  const int Kp   = K >> 2;      // packed bytes per output row

  v8f acc[2][2] = {};

  for (int kb = 0; kb < K; kb += BK) {
    const int blk = kb >> 7;
    // ---- kick off TDM DMA of the packed weight tile into LDS (wave 0 only)
#if HAVE_TDM
    if (wave == 0) {
      const uint64_t gaddr =
          (uint64_t)(uintptr_t)(Wp + (size_t)bn0 * Kp + (size_t)blk * 32);
      const uint32_t ldsa = (uint32_t)(uintptr_t)&Wstage[0];
      v4u g0 = {};
      v8i g1 = {};
      v4i gz = {};
      g0[0] = 1u;                                   // count=1, user descriptor
      g0[1] = ldsa;                                 // LDS destination address
      g0[2] = (uint32_t)gaddr;                      // global_addr[31:0]
      g0[3] = (uint32_t)((gaddr >> 32) & 0x01FFFFFFu) | (2u << 30); // [56:32]|type=2
      g1[0] = 0;                                    // wg_mask=0, data_size=1B
      g1[1] = (Kp & 0xFFFF) << 16;                  // tensor_dim0[15:0] @bit48
      g1[2] = ((Kp >> 16) & 0xFFFF) | ((N & 0xFFFF) << 16);  // dim0 hi | dim1 lo
      g1[3] = ((N >> 16) & 0xFFFF) | (32 << 16);    // dim1 hi | tile_dim0=32
      g1[4] = 64;                                   // tile_dim1=64, tile_dim2=0
      g1[5] = Kp;                                   // dim0_stride[31:0]
      g1[6] = 0;
      g1[7] = 0;
#if __clang_major__ >= 23
      v8i gz8 = {};
      __builtin_amdgcn_tensor_load_to_lds(g0, g1, gz, gz, gz8, 0);
#else
      __builtin_amdgcn_tensor_load_to_lds(g0, g1, gz, gz, 0);
#endif
    }
#else
    // fallback: plain coalesced copy of the packed tile into LDS
    for (int i = tid; i < BN * 8; i += 256) {
      const int n = i >> 3, w4 = i & 7;
      ((uint32_t*)Wstage)[n * 8 + w4] =
          *(const uint32_t*)(Wp + (size_t)(bn0 + n) * Kp + (size_t)blk * 32 + w4 * 4);
    }
#endif

    // ---- stage A tile: 8 independent float4 loads, then convert+store
    {
      const int r0 = tid >> 4;            // starting row (0..15)
      const int c  = (tid & 15) << 2;     // column (float4 granularity)
      const float* xp = &X[(size_t)(bm0 + r0) * K + kb + c];
      v4f x[8];
#pragma unroll
      for (int it = 0; it < 8; ++it)
        x[it] = *(const v4f*)(xp + (size_t)it * 16 * K);
#pragma unroll
      for (int it = 0; it < 8; ++it)
        *(v4bf*)&As[r0 + it * 16][c] = __builtin_convertvector(x[it], v4bf);
    }
    if (kb + BK < K)  // warm next A tile into cache
      __builtin_prefetch(&X[(size_t)(bm0 + (tid >> 1)) * K + kb + BK], 0, 1);

#if HAVE_TDM
    if (wave == 0) __builtin_amdgcn_s_wait_tensorcnt(0);
#endif
    __syncthreads();   // As staged, Wstage bytes landed

    // ---- decode ternary weights from LDS bytes (transposed store Bs[k][n]).
    // kb is 64-aligned -> window covers fields {f0, f0+1}, f0 in {0,2}.
    // Thread (j, n0): 8 strided byte reads, 2 ds_store_b128 (rows j, j+32).
    {
      const int f0    = (kb & 127) >> 5;
      const int sh_lo = 6 - 2 * f0;     // field for k-row j
      const int sh_hi = 4 - 2 * f0;     // field for k-row j+32
      const int j  = tid & 31;
      const int n0 = (tid >> 5) << 3;   // 0,8,...,56
      uint32_t wlo[4], whi[4];
#pragma unroll
      for (int t = 0; t < 4; ++t) {
        const uint32_t a0 = Wstage[(n0 + 2 * t) * 32 + j];
        const uint32_t a1 = Wstage[(n0 + 2 * t + 1) * 32 + j];
        wlo[t] = bfpat((a0 >> sh_lo) & 3) | (bfpat((a1 >> sh_lo) & 3) << 16);
        whi[t] = bfpat((a0 >> sh_hi) & 3) | (bfpat((a1 >> sh_hi) & 3) << 16);
      }
      v4u plo = {wlo[0], wlo[1], wlo[2], wlo[3]};
      v4u phi = {whi[0], whi[1], whi[2], whi[3]};
      *(v4u*)&Bs[j][n0]      = plo;
      *(v4u*)&Bs[j + 32][n0] = phi;
    }
    __syncthreads();   // Bs ready

#pragma unroll
    for (int kk = 0; kk < BK; kk += 32) {
      v16bf a[2], b[2];
#pragma unroll
      for (int i = 0; i < 2; ++i) {
        const int row = wm * 32 + i * 16 + ml;
        const v8bf lo = *(const v8bf*)&As[row][kk + 8 * half];
        const v8bf hi = *(const v8bf*)&As[row][kk + 16 + 8 * half];
        a[i] = cat8(lo, hi);
      }
#pragma unroll
      for (int j = 0; j < 2; ++j) {
#pragma unroll
        for (int s = 0; s < 16; ++s)
          b[j][s] = Bs[kk + (s < 8 ? 8 * half + s : 16 + 8 * half + (s - 8))][wn * 32 + j * 16 + ml];
      }
#pragma unroll
      for (int i = 0; i < 2; ++i)
#pragma unroll
        for (int j = 0; j < 2; ++j)
          acc[i][j] = __builtin_amdgcn_wmma_f32_16x16x32_bf16(
              false, a[i], false, b[j], (short)0, acc[i][j], false, false);
    }
    __syncthreads();   // done with As/Bs before next iteration overwrites
  }

  const float scale = scale_p[0];
#pragma unroll
  for (int i = 0; i < 2; ++i)
#pragma unroll
    for (int j = 0; j < 2; ++j)
#pragma unroll
      for (int r = 0; r < 8; ++r) {
        const int row = bm0 + wm * 32 + i * 16 + r + 8 * half;
        const int col = bn0 + wn * 32 + j * 16 + ml;
        Y[(size_t)row * N + col] = acc[i][j][r] * scale;
      }
}

// ---------------------------------------------------------------------------
// RoPE + repack to bf16 attention layouts.
//   Qr: (b, h, s, d)     Kt: (b, kvh, s, d)
//   Vt: (b, kvh, d, s) [transposed for B-fragments]
// ---------------------------------------------------------------------------
__global__ __launch_bounds__(128)
void rope_pack_kernel(const float* __restrict__ Q, const float* __restrict__ K,
                      const float* __restrict__ V, bf16_t* __restrict__ Qr,
                      bf16_t* __restrict__ Kt, bf16_t* __restrict__ Vt)
{
  const int tok = blockIdx.x;
  const int b   = tok / S_;
  const int s   = tok % S_;
  const int tid = threadIdx.x;
  const float LOG_THETA = 13.122363377404328f;  // ln(500000)

  for (int i = tid; i < NH_ * 64; i += 128) {
    const int h = i >> 6, d = i & 63;
    const float invf = __expf(-((float)d / 64.f) * LOG_THETA);
    float sn, cs;
    __sincosf((float)s * invf, &sn, &cs);
    const float x1 = Q[(size_t)tok * (NH_ * HD_) + h * HD_ + d];
    const float x2 = Q[(size_t)tok * (NH_ * HD_) + h * HD_ + d + 64];
    const size_t o = ((size_t)(b * NH_ + h) * S_ + s) * HD_;
    Qr[o + d]      = (bf16_t)(x1 * cs - x2 * sn);
    Qr[o + d + 64] = (bf16_t)(x2 * cs + x1 * sn);
  }
  for (int i = tid; i < NKV_ * 64; i += 128) {
    const int h = i >> 6, d = i & 63;
    const float invf = __expf(-((float)d / 64.f) * LOG_THETA);
    float sn, cs;
    __sincosf((float)s * invf, &sn, &cs);
    const float x1 = K[(size_t)tok * (NKV_ * HD_) + h * HD_ + d];
    const float x2 = K[(size_t)tok * (NKV_ * HD_) + h * HD_ + d + 64];
    const size_t o = ((size_t)(b * NKV_ + h) * S_ + s) * HD_;
    Kt[o + d]      = (bf16_t)(x1 * cs - x2 * sn);
    Kt[o + d + 64] = (bf16_t)(x2 * cs + x1 * sn);
  }
  for (int i = tid; i < NKV_ * HD_; i += 128) {
    const int h = i >> 7, d = i & 127;
    Vt[(size_t)(b * NKV_ + h) * HD_ * S_ + (size_t)d * S_ + s] =
        (bf16_t)V[(size_t)tok * (NKV_ * HD_) + h * HD_ + d];
  }
}

// ---------------------------------------------------------------------------
// Causal flash attention, one (b,h) x 64 q-rows per block; 4 waves, each wave
// owns 16 q-rows. Online softmax operates directly in C-fragment layout:
// row = acc-VGPR index, so per-row rescale is a plain scalar multiply.
// All K/V/Q fragments are two 16-byte per-lane vector loads.
// ---------------------------------------------------------------------------
__global__ __launch_bounds__(128)
void flash_attn_kernel(const bf16_t* __restrict__ Qr, const bf16_t* __restrict__ Kt,
                       const bf16_t* __restrict__ Vt, float* __restrict__ Out)
{
  __shared__ bf16_t Pbuf[4][16][40];   // per-wave P staging; row stride 80 B

  const int tid  = threadIdx.x;
  const int lane = tid & 31;
  const int wave = tid >> 5;
  const int half = lane >> 4;
  const int ml   = lane & 15;

  const int bh   = blockIdx.y;
  const int b    = bh / NH_;
  const int h    = bh % NH_;
  const int kvh  = h / GROUPS_;
  const int qbase = blockIdx.x * 64 + wave * 16;
  const int qend  = blockIdx.x * 64 + 64;   // uniform causal bound for block

  const size_t qoff = (size_t)(b * NH_ + h) * S_ * HD_;
  const size_t koff = (size_t)(b * NKV_ + kvh) * HD_ * S_;
  const size_t voff = (size_t)(b * NKV_ + kvh) * S_ * HD_;

  // Preload this wave's 16x128 Q block as four 16x32 A-fragments (vector loads).
  v16bf aQ[4];
  {
    const bf16_t* qr = Qr + qoff + (size_t)(qbase + ml) * HD_;
#pragma unroll
    for (int c = 0; c < 4; ++c) {
      const v8bf lo = *(const v8bf*)(qr + c * 32 + 8 * half);
      const v8bf hi = *(const v8bf*)(qr + c * 32 + 16 + 8 * half);
      aQ[c] = cat8(lo, hi);
    }
  }

  v8f oacc[8] = {};
  float mrow[8], lrow[8];
#pragma unroll
  for (int r = 0; r < 8; ++r) { mrow[r] = -__builtin_inff(); lrow[r] = 0.f; }

  const float sm_scale = 0.08838834764831845f;   // 1/sqrt(128)

  for (int kb = 0; kb < qend; kb += 32) {
    // ---- scores: S(16x32) = Q(16x128) . K^T(128x32), 8 WMMAs
    v8f sc0 = {}, sc1 = {};
#pragma unroll
    for (int c = 0; c < 4; ++c) {
      // lane = key col; d in table order -> two 16-byte loads per fragment
      const bf16_t* kp = Kt + koff + (size_t)(kb + ml) * HD_ + c * 32;
      const v16bf bK0 = cat8(*(const v8bf*)(kp + 8 * half), *(const v8bf*)(kp + 16 + 8 * half));
      const v16bf bK1 = cat8(*(const v8bf*)(kp + 16 * HD_ + 8 * half), *(const v8bf*)(kp + 16 * HD_ + 16 + 8 * half));
      sc0 = __builtin_amdgcn_wmma_f32_16x16x32_bf16(false, aQ[c], false, bK0,
                                                    (short)0, sc0, false, false);
      sc1 = __builtin_amdgcn_wmma_f32_16x16x32_bf16(false, aQ[c], false, bK1,
                                                    (short)0, sc1, false, false);
    }

    // ---- online softmax in C layout (lane = key col, VGPR r = q row)
    float p0[8], p1[8], alpha[8];
#pragma unroll
    for (int r = 0; r < 8; ++r) {
      const int qg  = qbase + r + 8 * half;
      const int kg0 = kb + ml;
      const int kg1 = kb + 16 + ml;
      const float s0 = (kg0 <= qg) ? sc0[r] * sm_scale : -3.0e38f;
      const float s1 = (kg1 <= qg) ? sc1[r] * sm_scale : -3.0e38f;
      float mx = fmaxf(s0, s1);
#pragma unroll
      for (int off = 8; off >= 1; off >>= 1)   // 16-lane butterfly (per half)
        mx = fmaxf(mx, __shfl_xor(mx, off));
      const float mnew = fmaxf(mrow[r], mx);
      alpha[r] = __expf(mrow[r] - mnew);
      const float e0 = (kg0 <= qg) ? __expf(s0 - mnew) : 0.f;
      const float e1 = (kg1 <= qg) ? __expf(s1 - mnew) : 0.f;
      float sum = e0 + e1;
#pragma unroll
      for (int off = 8; off >= 1; off >>= 1)
        sum += __shfl_xor(sum, off);
      lrow[r] = lrow[r] * alpha[r] + sum;
      mrow[r] = mnew;
      p0[r] = e0;
      p1[r] = e1;
    }
#pragma unroll
    for (int t = 0; t < 8; ++t)
#pragma unroll
      for (int r = 0; r < 8; ++r)
        oacc[t][r] *= alpha[r];

    // ---- C-layout -> A-layout transpose of P through per-wave LDS
#pragma unroll
    for (int r = 0; r < 8; ++r) {
      Pbuf[wave][r + 8 * half][ml]      = (bf16_t)p0[r];
      Pbuf[wave][r + 8 * half][16 + ml] = (bf16_t)p1[r];
    }
    asm volatile("s_wait_dscnt 0" ::: "memory");   // per-wave DS RAW fence
    const v8bf plo = *(const v8bf*)&Pbuf[wave][ml][8 * half];
    const v8bf phi = *(const v8bf*)&Pbuf[wave][ml][16 + 8 * half];
    const v16bf aP = cat8(plo, phi);
    asm volatile("s_wait_dscnt 0" ::: "memory");   // loads done before reuse

    // ---- O += P(16x32) . V(32x128), 8 WMMAs
#pragma unroll
    for (int t = 0; t < 8; ++t) {
      // lane = d col; keys in table order -> two 16-byte loads
      const bf16_t* vp = Vt + voff + (size_t)(t * 16 + ml) * S_ + kb;
      const v16bf bV = cat8(*(const v8bf*)(vp + 8 * half), *(const v8bf*)(vp + 16 + 8 * half));
      oacc[t] = __builtin_amdgcn_wmma_f32_16x16x32_bf16(false, aP, false, bV,
                                                        (short)0, oacc[t], false, false);
    }
  }

  // ---- epilogue: normalize by l and write (b, s, h*128 + d) f32
#pragma unroll
  for (int r = 0; r < 8; ++r) {
    const float inv = 1.f / lrow[r];
    const int sg = qbase + r + 8 * half;
    const size_t base = ((size_t)b * S_ + sg) * (size_t)(NH_ * HD_) + h * HD_;
#pragma unroll
    for (int t = 0; t < 8; ++t)
      Out[base + t * 16 + ml] = oacc[t][r] * inv;
  }
}

// ---------------------------------------------------------------------------
// RMSNorm over 2560 features per token (single HBM read via LDS cache).
// ---------------------------------------------------------------------------
__global__ __launch_bounds__(256)
void rmsnorm_kernel(const float* __restrict__ X, const float* __restrict__ w,
                    float* __restrict__ Y)
{
  __shared__ float xs[HID_];
  __shared__ float red[8];
  const int tok = blockIdx.x;
  const int tid = threadIdx.x;
  const size_t base = (size_t)tok * HID_;

  float ss = 0.f;
  for (int i = tid; i < HID_; i += 256) {
    const float v = X[base + i];
    xs[i] = v;
    ss += v * v;
  }
#pragma unroll
  for (int off = 16; off >= 1; off >>= 1) ss += __shfl_xor(ss, off);
  if ((tid & 31) == 0) red[tid >> 5] = ss;
  __syncthreads();
  if (tid == 0) {
    float t = 0.f;
#pragma unroll
    for (int i = 0; i < 8; ++i) t += red[i];
    red[0] = t;
  }
  __syncthreads();
  const float inv = rsqrtf(red[0] / (float)HID_ + 1e-6f);
  for (int i = tid; i < HID_; i += 256)
    Y[base + i] = xs[i] * inv * w[i];
}

// ---------------------------------------------------------------------------
extern "C" void kernel_launch(void* const* d_in, const int* in_sizes, int n_in,
                              void* d_out, int out_size, void* d_ws, size_t ws_size,
                              hipStream_t stream) {
  (void)in_sizes; (void)n_in; (void)out_size; (void)ws_size;

  const float*   hs     = (const float*)d_in[0];
  // d_in[1] = attention_mask: exactly causal in the reference; applied inline.
  const uint8_t* q_w    = (const uint8_t*)d_in[2];
  const uint8_t* k_w    = (const uint8_t*)d_in[3];
  const uint8_t* v_w    = (const uint8_t*)d_in[4];
  const uint8_t* o_w    = (const uint8_t*)d_in[5];
  const float*   q_s    = (const float*)d_in[6];
  const float*   k_s    = (const float*)d_in[7];
  const float*   v_s    = (const float*)d_in[8];
  const float*   o_s    = (const float*)d_in[9];
  const float*   norm_w = (const float*)d_in[10];
  float*         out    = (float*)d_out;

  char* ws = (char*)d_ws;
  size_t off = 0;
  auto take = [&](size_t bytes) -> char* {
    char* p = ws + off;
    off += (bytes + 255) & ~(size_t)255;
    return p;
  };
  float*  qbuf   = (float*)take((size_t)TOKENS_ * 2560 * 4);
  float*  kbuf   = (float*)take((size_t)TOKENS_ * 640 * 4);
  float*  vbuf   = (float*)take((size_t)TOKENS_ * 640 * 4);
  bf16_t* Qr     = (bf16_t*)take((size_t)TOKENS_ * 2560 * 2);
  bf16_t* Kt     = (bf16_t*)take((size_t)TOKENS_ * 640 * 2);
  bf16_t* Vt     = (bf16_t*)take((size_t)TOKENS_ * 640 * 2);
  float*  attn   = (float*)take((size_t)TOKENS_ * 2560 * 4);
  float*  normed = (float*)take((size_t)TOKENS_ * 2560 * 4);

  // 1) ternary projections (WMMA GEMMs, TDM weight staging)
  bitlinear_wmma<<<dim3(2560 / BN, TOKENS_ / BM), 256, 0, stream>>>(
      hs, q_w, q_s, qbuf, TOKENS_, HID_, 2560);
  bitlinear_wmma<<<dim3(640 / BN, TOKENS_ / BM), 256, 0, stream>>>(
      hs, k_w, k_s, kbuf, TOKENS_, HID_, 640);
  bitlinear_wmma<<<dim3(640 / BN, TOKENS_ / BM), 256, 0, stream>>>(
      hs, v_w, v_s, vbuf, TOKENS_, HID_, 640);
  // 2) RoPE + bf16 repack
  rope_pack_kernel<<<TOKENS_, 128, 0, stream>>>(qbuf, kbuf, vbuf, Qr, Kt, Vt);
  // 3) causal flash attention (WMMA)
  flash_attn_kernel<<<dim3(S_ / 64, B_ * NH_), 128, 0, stream>>>(Qr, Kt, Vt, attn);
  // 4) RMSNorm
  rmsnorm_kernel<<<TOKENS_, 256, 0, stream>>>(attn, norm_w, normed);
  // 5) output ternary projection (WMMA GEMM)
  bitlinear_wmma<<<dim3(2560 / BN, TOKENS_ / BM), 256, 0, stream>>>(
      normed, o_w, o_s, out, TOKENS_, NH_ * HD_, 2560);
}
